// GCN_Critic_24223615550522
// MI455X (gfx1250) — matrix operand order repaired and checked by reading, not compiled, not run on hardware
//
#include <hip/hip_runtime.h>
#include <hip/hip_bf16.h>

typedef __attribute__((ext_vector_type(16))) __bf16 v16bf;
typedef __attribute__((ext_vector_type(8)))  float  v8f;

#define HD 128   // feature / hidden dim (fixed by the model)

// ---------------------------------------------------------------- utilities
__global__ void fill_kernel(float* p, float v, int n) {
    int i = blockIdx.x * blockDim.x + threadIdx.x;
    if (i < n) p[i] = v;
}

// deg[dst] += ew[e]   (deg pre-filled with 1.0 for the self loop)
__global__ void deg_accum_kernel(const int* ei, const float* ew, float* deg, int E) {
    int e = blockIdx.x * blockDim.x + threadIdx.x;
    if (e < E) atomicAdd(&deg[ei[E + e]], ew[e]);
}

__global__ void rsqrt_kernel(float* d, int n) {
    int i = blockIdx.x * blockDim.x + threadIdx.x;
    if (i < n) {
        float v = d[i];
        d[i] = (v > 0.0f) ? rsqrtf(v) : 0.0f;
    }
}

// ---------------------------------------------------- W pack (B fragments)
// Bp[((ct*4 + ks)*32 + lane)*16 + i] = bf16( W[(ks*32 + (lane<16?0:16) + i)][ct*16 + (lane&15)] )
__global__ void packW_kernel(const float* W, __bf16* Bp) {
    int t = blockIdx.x * blockDim.x + threadIdx.x;
    if (t >= 1024) return;
    int ct   = t >> 7;
    int ks   = (t >> 5) & 3;
    int lane = t & 31;
    int kbase = ks * 32 + ((lane < 16) ? 0 : 16);
    int n     = ct * 16 + (lane & 15);
    __bf16* dst = Bp + (size_t)((ct * 4 + ks) * 32 + lane) * 16;
#pragma unroll
    for (int i = 0; i < 16; ++i) dst[i] = (__bf16)W[(size_t)(kbase + i) * HD + n];
}

// ------------------------------------------------------------- WMMA GEMM
// h[N x 128] = x[N x 128] @ W[128 x 128]; block = 8 waves, wave w = col tile w.
__global__ void gemm_wmma_kernel(const float* __restrict__ x,
                                 const __bf16* __restrict__ Bp,
                                 float* __restrict__ h, int N) {
    int lane = threadIdx.x & 31;
    int ct   = threadIdx.x >> 5;            // 0..7 : 16-column tile
    int row0 = blockIdx.x * 16;
    int m    = row0 + (lane & 15);
    if (m >= N) m = N - 1;
    const float* xr = x + (size_t)m * HD;
    int kl = (lane < 16) ? 0 : 16;

    v8f c = {};
#pragma unroll
    for (int ks = 0; ks < 4; ++ks) {
        int kk = ks * 32;
        // A fragment: 16 bf16 per lane; K = kk+kl..kk+kl+15
        const float4* p0 = reinterpret_cast<const float4*>(xr + kk + kl);
        const float4* p1 = reinterpret_cast<const float4*>(xr + kk + kl + 8);
        float fa[16];
        *reinterpret_cast<float4*>(fa + 0)  = p0[0];
        *reinterpret_cast<float4*>(fa + 4)  = p0[1];
        *reinterpret_cast<float4*>(fa + 8)  = p1[0];
        *reinterpret_cast<float4*>(fa + 12) = p1[1];
        v16bf a;
#pragma unroll
        for (int i = 0; i < 16; ++i) a[i] = (__bf16)fa[i];

        v16bf b = *reinterpret_cast<const v16bf*>(
            Bp + (size_t)((ct * 4 + ks) * 32 + lane) * 16);

        c = __builtin_amdgcn_wmma_f32_16x16x32_bf16(
            false, a, false, b, (short)0, c, false, false);
    }

    int rbase = (lane < 16) ? 0 : 8;
    int col   = ct * 16 + (lane & 15);
#pragma unroll
    for (int r = 0; r < 8; ++r) {
        int row = row0 + rbase + r;
        if (row < N) h[(size_t)row * HD + col] = c[r];
    }
}

// ---------------------------------------- self loop (also initializes agg)
__global__ void selfloop_kernel(const float* __restrict__ h,
                                const float* __restrict__ dinv,
                                float* __restrict__ agg, int N) {
    long t = (long)blockIdx.x * blockDim.x + threadIdx.x;
    long node = t >> 5;
    int lane  = (int)(t & 31);
    if (node >= N) return;
    float d = dinv[node];
    float s = d * d;
    float4 v = reinterpret_cast<const float4*>(h + node * HD)[lane];
    float4 o = make_float4(v.x * s, v.y * s, v.z * s, v.w * s);
    reinterpret_cast<float4*>(agg + node * HD)[lane] = o;
}

// ------------------------------------------------- edge scatter (1 wave/edge)
__global__ void scatter_kernel(const float* __restrict__ h,
                               const int* __restrict__ ei,
                               const float* __restrict__ ew,
                               const float* __restrict__ dinv,
                               float* __restrict__ agg, int E) {
    long t = (long)blockIdx.x * blockDim.x + threadIdx.x;
    long e = t >> 5;
    int lane = (int)(t & 31);
    if (e >= E) return;
    int src = ei[e];
    int dst = ei[(long)E + e];
    float norm = dinv[src] * ew[e] * dinv[dst];
    float4 v = reinterpret_cast<const float4*>(h + (size_t)src * HD)[lane];
    float* o = agg + (size_t)dst * HD + lane * 4;
    atomicAdd(o + 0, v.x * norm);
    atomicAdd(o + 1, v.y * norm);
    atomicAdd(o + 2, v.z * norm);
    atomicAdd(o + 3, v.w * norm);
}

// ---------------- relu + bias + l2norm + mean-pool partials (64 nodes/block)
__global__ void post_kernel(const float* __restrict__ agg,
                            const float* __restrict__ bias,
                            float* __restrict__ mean_acc, int N) {
    int wave = threadIdx.x >> 5;
    int lane = threadIdx.x & 31;
    int nodeBase = blockIdx.x * 64 + wave * 8;
    float4 bv  = reinterpret_cast<const float4*>(bias)[lane];
    float4 acc = make_float4(0.f, 0.f, 0.f, 0.f);
    for (int i = 0; i < 8; ++i) {
        int node = nodeBase + i;
        if (node < N) {
            float4 v = reinterpret_cast<const float4*>(agg + (size_t)node * HD)[lane];
            v.x = fmaxf(v.x + bv.x, 0.f);
            v.y = fmaxf(v.y + bv.y, 0.f);
            v.z = fmaxf(v.z + bv.z, 0.f);
            v.w = fmaxf(v.w + bv.w, 0.f);
            float s = v.x * v.x + v.y * v.y + v.z * v.z + v.w * v.w;
            for (int off = 16; off > 0; off >>= 1) s += __shfl_xor(s, off, 32);
            float sc = 1.0f / fmaxf(sqrtf(s), 1e-12f);
            acc.x += v.x * sc; acc.y += v.y * sc;
            acc.z += v.z * sc; acc.w += v.w * sc;
        }
    }
    __shared__ float4 lds[8][32];
    lds[wave][lane] = acc;
    __syncthreads();
    if (wave == 0) {
        float4 tsum = lds[0][lane];
#pragma unroll
        for (int w = 1; w < 8; ++w) {
            tsum.x += lds[w][lane].x; tsum.y += lds[w][lane].y;
            tsum.z += lds[w][lane].z; tsum.w += lds[w][lane].w;
        }
        float* d = mean_acc + lane * 4;
        atomicAdd(d + 0, tsum.x);
        atomicAdd(d + 1, tsum.y);
        atomicAdd(d + 2, tsum.z);
        atomicAdd(d + 3, tsum.w);
    }
}

// ------------------------------------------------------------- head MLP
__global__ void mlp_kernel(const float* __restrict__ mean_acc,
                           float invN_net, float invN_dag,
                           const float* __restrict__ W1, const float* __restrict__ b1,
                           const float* __restrict__ W2, const float* __restrict__ b2,
                           float* __restrict__ out) {
    __shared__ float comb[256];
    __shared__ float red[64];
    int t = threadIdx.x;
    for (int k = t; k < 256; k += 64)
        comb[k] = mean_acc[k] * ((k < HD) ? invN_net : invN_dag);
    __syncthreads();
    float hj = b1[t];
    for (int k = 0; k < 256; ++k) hj += comb[k] * W1[k * 64 + t];
    hj = fmaxf(hj, 0.f);
    red[t] = hj * W2[t];
    __syncthreads();
    for (int s = 32; s > 0; s >>= 1) {
        if (t < s) red[t] += red[t + s];
        __syncthreads();
    }
    if (t == 0) out[0] = red[0] + b2[0];
}

// ---------------------------------------------------------------- launch
extern "C" void kernel_launch(void* const* d_in, const int* in_sizes, int n_in,
                              void* d_out, int out_size, void* d_ws, size_t ws_size,
                              hipStream_t stream) {
    const float* net_feat = (const float*)d_in[0];
    const int*   net_ei   = (const int*)d_in[1];
    const float* net_ew   = (const float*)d_in[2];
    const float* dag_feat = (const float*)d_in[3];
    const int*   dag_ei   = (const int*)d_in[4];
    const float* dag_ew   = (const float*)d_in[5];
    const float* W_net    = (const float*)d_in[6];
    const float* b_net    = (const float*)d_in[7];
    const float* W_dag    = (const float*)d_in[8];
    const float* b_dag    = (const float*)d_in[9];
    const float* W1       = (const float*)d_in[10];
    const float* b1       = (const float*)d_in[11];
    const float* W2       = (const float*)d_in[12];
    const float* b2       = (const float*)d_in[13];
    float* out = (float*)d_out;

    const int Nn = in_sizes[0] / HD;
    const int En = in_sizes[2];
    const int Nd = in_sizes[3] / HD;
    const int Ed = in_sizes[5];

    // workspace carve-up (256B aligned)
    char* ws = (char*)d_ws;
    size_t off = 0;
    auto take = [&](size_t bytes) {
        void* p = ws + off;
        off = (off + bytes + 255) & ~(size_t)255;
        return p;
    };
    float*  h_net    = (float*)take((size_t)Nn * HD * 4);
    float*  agg_net  = (float*)take((size_t)Nn * HD * 4);
    float*  h_dag    = (float*)take((size_t)Nd * HD * 4);
    float*  agg_dag  = (float*)take((size_t)Nd * HD * 4);
    float*  dinv_net = (float*)take((size_t)Nn * 4);
    float*  dinv_dag = (float*)take((size_t)Nd * 4);
    __bf16* Bp_net   = (__bf16*)take(8 * 4 * 32 * 16 * 2);
    __bf16* Bp_dag   = (__bf16*)take(8 * 4 * 32 * 16 * 2);
    float*  mean_acc = (float*)take(256 * 4);
    (void)ws_size; (void)n_in; (void)out_size;

    const int B = 256;
    auto blk = [](long n, int b) { return (int)((n + b - 1) / b); };

    // 1. degree init (self-loop weight 1.0), mean accumulator zero
    fill_kernel<<<blk(Nn, B), B, 0, stream>>>(dinv_net, 1.0f, Nn);
    fill_kernel<<<blk(Nd, B), B, 0, stream>>>(dinv_dag, 1.0f, Nd);
    fill_kernel<<<1, 256, 0, stream>>>(mean_acc, 0.0f, 256);

    // 2. degree accumulate + rsqrt
    deg_accum_kernel<<<blk(En, B), B, 0, stream>>>(net_ei, net_ew, dinv_net, En);
    deg_accum_kernel<<<blk(Ed, B), B, 0, stream>>>(dag_ei, dag_ew, dinv_dag, Ed);
    rsqrt_kernel<<<blk(Nn, B), B, 0, stream>>>(dinv_net, Nn);
    rsqrt_kernel<<<blk(Nd, B), B, 0, stream>>>(dinv_dag, Nd);

    // 3. pack weights into WMMA B-fragment order
    packW_kernel<<<4, 256, 0, stream>>>(W_net, Bp_net);
    packW_kernel<<<4, 256, 0, stream>>>(W_dag, Bp_dag);

    // 4. h = x @ W via v_wmma_f32_16x16x32_bf16
    gemm_wmma_kernel<<<blk(Nn, 16), 256, 0, stream>>>(net_feat, Bp_net, h_net, Nn);
    gemm_wmma_kernel<<<blk(Nd, 16), 256, 0, stream>>>(dag_feat, Bp_dag, h_dag, Nd);

    // 5. self-loop contribution (initializes agg)
    selfloop_kernel<<<blk((long)Nn * 32, B), B, 0, stream>>>(h_net, dinv_net, agg_net, Nn);
    selfloop_kernel<<<blk((long)Nd * 32, B), B, 0, stream>>>(h_dag, dinv_dag, agg_dag, Nd);

    // 6. edge scatter-add (one wave per edge, float4 lanes)
    scatter_kernel<<<blk((long)En * 32, B), B, 0, stream>>>(h_net, net_ei, net_ew, dinv_net, agg_net, En);
    scatter_kernel<<<blk((long)Ed * 32, B), B, 0, stream>>>(h_dag, dag_ei, dag_ew, dinv_dag, agg_dag, Ed);

    // 7. relu + bias + l2norm + mean partials
    post_kernel<<<blk(Nn, 64), 256, 0, stream>>>(agg_net, b_net, mean_acc, Nn);
    post_kernel<<<blk(Nd, 64), 256, 0, stream>>>(agg_dag, b_dag, mean_acc + HD, Nd);

    // 8. head MLP -> scalar
    mlp_kernel<<<1, 64, 0, stream>>>(mean_acc, 1.0f / (float)Nn, 1.0f / (float)Nd,
                                     W1, b1, W2, b2, out);
}
